// Attention_47090021433597
// MI455X (gfx1250) — matrix operand order repaired and checked by reading, not compiled, not run on hardware
//
#include <hip/hip_runtime.h>
#include <cmath>

// ---------------- CDNA5 WMMA types / helpers ----------------
typedef __attribute__((ext_vector_type(16))) __bf16 v16bf;
typedef __attribute__((ext_vector_type(8)))  float  v8f;
typedef __attribute__((ext_vector_type(8)))  unsigned short us8;

union BFrag { v16bf v; unsigned short u[16]; us8 h[2]; };

__device__ __forceinline__ unsigned short f2bf(float f) {
  __bf16 h = (__bf16)f;                       // native v_cvt to bf16
  return __builtin_bit_cast(unsigned short, h);
}

__device__ __forceinline__ v8f wmma_bf16(v16bf a, v16bf b, v8f c) {
  return __builtin_amdgcn_wmma_f32_16x16x32_bf16(false, a, false, b, (short)0, c, false, false);
}

// A fragment: 16x32 bf16, row-major source [m][k], contiguous k order (B loaders match):
// lane L: m = L&15, 16 consecutive K starting at k0 + 16*(L>>4).
// Requires (m0? no) row base 16B aligned: stride%8==0 and k0%8==0 (true for all call sites).
__device__ __forceinline__ v16bf load_a(const unsigned short* base, int stride,
                                        int m0, int k0, int lane) {
  BFrag f;
  const unsigned short* row = base + (m0 + (lane & 15)) * stride + k0 + ((lane >> 4) << 4);
  f.h[0] = *(const us8*)(row);
  f.h[1] = *(const us8*)(row + 8);
  return f.v;
}

// B fragment (32x16) from B-transposed storage, row-major [n][k] (e.g. weight[o][c]):
// lane L: n = L&15, 16 consecutive K starting at k0 + 16*(L>>4).
__device__ __forceinline__ v16bf load_bt(const unsigned short* base, int stride,
                                         int k0, int n0, int lane) {
  BFrag f;
  const unsigned short* row = base + (n0 + (lane & 15)) * stride + k0 + ((lane >> 4) << 4);
  f.h[0] = *(const us8*)(row);
  f.h[1] = *(const us8*)(row + 8);
  return f.v;
}

// B fragment (32x16) from row-major [k][n] storage (e.g. V matrix rows = K dim).
__device__ __forceinline__ v16bf load_b(const unsigned short* base, int stride,
                                        int k0, int n0, int lane) {
  BFrag f;
  const unsigned short* p = base + (k0 + ((lane >> 4) << 4)) * stride + n0 + (lane & 15);
#pragma unroll
  for (int i = 0; i < 16; ++i) f.u[i] = p[i * stride];
  return f.v;
}

// ---------------- problem constants ----------------
#define BATCH   4
#define C_DIM   96
#define HWDIM   256
#define SCALE_F 0.17677669529663687f   // (96/3)^-0.5

// LDS strides (elements) -- all %8==0 so fragment rows are 16B aligned
#define XS  104   // x window rows [pixel][ch]
#define QS  488   // qkv rows [pixel][480 cols]
#define AOS 104   // attn-out rows [pixel][96]
#define PS  72    // P rows [row][64]
#define SS  65    // S rows (float)

// ws element counts
#define WM_ELE (288 * 96)
#define WI_ELE (192 * 96)
#define WP_ELE (96 * 96)

// ---------------- prep: pack weights to bf16 + conv-bias vector ----------------
__global__ void pack_kernel(const float* V_w, const float* V_b,
                            const float* Vm_w, const float* Vm_b,
                            const float* QK_w, const float* QK_b,
                            const float* QKm_w, const float* QKm_b,
                            const float* proj_w, const float* projm_w,
                            unsigned short* wm, unsigned short* wi,
                            unsigned short* wp, unsigned short* wpm, float* bv) {
  const int total = WM_ELE + WI_ELE + 2 * WP_ELE + 480;
  for (int e = blockIdx.x * blockDim.x + threadIdx.x; e < total;
       e += gridDim.x * blockDim.x) {
    int i = e;
    if (i < WM_ELE) {                       // [QKm_w ; Vm_w]  288x96
      int o = i / 96, k = i % 96;
      float v = (o < 192) ? QKm_w[o * 96 + k] : Vm_w[(o - 192) * 96 + k];
      wm[i] = f2bf(v);
      continue;
    }
    i -= WM_ELE;
    if (i < WI_ELE) {                       // [QK_w rows 96..191 ; V_w]  192x96
      int o = i / 96, k = i % 96;
      float v = (o < 96) ? QK_w[(96 + o) * 96 + k] : V_w[(o - 96) * 96 + k];
      wi[i] = f2bf(v);
      continue;
    }
    i -= WI_ELE;
    if (i < WP_ELE) { wp[i] = f2bf(proj_w[i]); continue; }
    i -= WP_ELE;
    if (i < WP_ELE) { wpm[i] = f2bf(projm_w[i]); continue; }
    i -= WP_ELE;
    // conv bias vector over the 480 fused output columns
    int c = i;
    float v;
    if (c < 192)      v = QKm_b[c];
    else if (c < 288) v = Vm_b[c - 192];
    else if (c < 384) v = QK_b[96 + (c - 288)];
    else              v = V_b[c - 384];
    bv[c] = v;
  }
}

// ---------------- prep: relative-position bias MLP  -> abias[3][64][64] ----------------
__global__ void bias_kernel(const float* w1, const float* b1,
                            const float* w2, const float* b2, float* abias) {
  int p = blockIdx.x * blockDim.x + threadIdx.x;
  if (p >= 4096) return;
  int n = p >> 6, m = p & 63;
  float d0 = (float)((n >> 3) - (m >> 3));
  float d1 = (float)((n & 7) - (m & 7));
  float s0 = (d0 > 0.f) ? 1.f : ((d0 < 0.f) ? -1.f : 0.f);
  float s1 = (d1 > 0.f) ? 1.f : ((d1 < 0.f) ? -1.f : 0.f);
  float r0 = s0 * log1pf(fabsf(d0));
  float r1 = s1 * log1pf(fabsf(d1));
  float a0 = b2[0], a1 = b2[1], a2 = b2[2];
  for (int j = 0; j < 256; ++j) {
    float h = fmaxf(r0 * w1[j] + r1 * w1[256 + j] + b1[j], 0.f);
    a0 += h * w2[j * 3 + 0];
    a1 += h * w2[j * 3 + 1];
    a2 += h * w2[j * 3 + 2];
  }
  abias[0 * 4096 + n * 64 + m] = a0;
  abias[1 * 4096 + n * 64 + m] = a1;
  abias[2 * 4096 + n * 64 + m] = a2;
}

// ---------------- fused per-window attention kernel ----------------
__global__ __launch_bounds__(256)
void attn_kernel(const float* __restrict__ x_ivif, const float* __restrict__ x_mfif,
                 const unsigned short* __restrict__ wm, const unsigned short* __restrict__ wi,
                 const unsigned short* __restrict__ wp, const unsigned short* __restrict__ wpm,
                 const float* __restrict__ bv_g, const float* __restrict__ abias,
                 const float* __restrict__ proj_b, const float* __restrict__ projm_b,
                 float* __restrict__ out_i, float* __restrict__ out_m) {
  extern __shared__ char smem_raw[];
  unsigned short* ls_x   = (unsigned short*)smem_raw;          // [2][64][XS]  (0=mfif, 1=ivif)
  unsigned short* ls_qkv = ls_x + 2 * 64 * XS;                 // [64][QS]
  unsigned short* ls_ao  = ls_qkv + 64 * QS;                   // [2][64][AOS] (0=m branch, 1=i branch)
  unsigned short* ls_P   = ls_ao + 2 * 64 * AOS;               // [64][PS]
  float*          ls_S   = (float*)(ls_P + 64 * PS);           // [64][SS]
  float*          ls_bv  = ls_S + 64 * SS;                     // [480]
  float*          ls_red = ls_bv + 480;                        // [64][4] softmax scratch

  const int tid  = threadIdx.x;
  const int wv   = tid >> 5;
  const int lane = tid & 31;

  const int bid = blockIdx.x;
  const int b   = bid >> 10;
  const int wy  = (bid >> 5) & 31;
  const int wx  = bid & 31;
  const int h0  = wy * 8, w0 = wx * 8;

  // ---- phase 1: load both x windows as bf16 [pixel][channel] ----
  for (int e = tid; e < 480; e += 256) ls_bv[e] = bv_g[e];
  for (int e = tid; e < 2 * 64 * 96; e += 256) {
    int src = e / 6144;              // 0 = mfif, 1 = ivif
    int r   = e % 6144;
    int c   = r >> 6;
    int p   = r & 63;
    const float* xp = src ? x_ivif : x_mfif;
    float v = __builtin_nontemporal_load(
        xp + (((b * C_DIM + c) * HWDIM) + h0 + (p >> 3)) * HWDIM + w0 + (p & 7));
    ls_x[(src * 64 + p) * XS + c] = f2bf(v);
  }
  __syncthreads();

  // ---- phase 2: fused QKV GEMM  [64 x 480] = x @ W^T  (+bias, q scaled) ----
  // cols 0..95 qm*SCALE | 96..191 km | 192..287 vm | 288..383 k_i | 384..479 v_i
  for (int job = wv; job < 120; job += 8) {
    int mt = job / 30, nt = job % 30;
    int m0 = mt * 16, n0 = nt * 16;
    const unsigned short* xbase = ls_x + ((n0 >= 288) ? 64 * XS : 0);
    v8f acc = {0.f, 0.f, 0.f, 0.f, 0.f, 0.f, 0.f, 0.f};
#pragma unroll
    for (int ks = 0; ks < 3; ++ks) {
      v16bf a = load_a(xbase, XS, m0, ks * 32, lane);
      v16bf bb = (n0 < 288) ? load_bt(wm, 96, ks * 32, n0, lane)
                            : load_bt(wi, 96, ks * 32, n0 - 288, lane);
      acc = wmma_bf16(a, bb, acc);
    }
    int n = n0 + (lane & 15);
    float cb = ls_bv[n];
    float sc = (n < 96) ? SCALE_F : 1.0f;
#pragma unroll
    for (int j = 0; j < 8; ++j) {
      int m = m0 + j + ((lane >> 4) << 3);
      ls_qkv[m * QS + n] = f2bf((acc[j] + cb) * sc);
    }
  }
  __syncthreads();

  // ---- phases 3/4: two attentions (br 0: mfif k/v ; br 1: ivif k/v), q = qm*SCALE ----
  for (int br = 0; br < 2; ++br) {
    const int kcol = (br == 0) ? 96 : 288;
    const int vcol = (br == 0) ? 192 : 384;
    for (int hh = 0; hh < 3; ++hh) {
      const int qb = hh * 32, kb = kcol + hh * 32, vb = vcol + hh * 32;
      // scores: S[64x64] = q @ k^T + abias   (K = 32 -> one WMMA per tile)
      for (int job = wv; job < 16; job += 8) {
        int m0 = (job >> 2) * 16, n0 = (job & 3) * 16;
        v16bf a  = load_a(ls_qkv, QS, m0, qb, lane);
        v16bf bb = load_bt(ls_qkv, QS, kb, n0, lane);
        v8f z = {0.f, 0.f, 0.f, 0.f, 0.f, 0.f, 0.f, 0.f};
        v8f acc = wmma_bf16(a, bb, z);
        int n = n0 + (lane & 15);
        const float* ab = abias + hh * 4096;
#pragma unroll
        for (int j = 0; j < 8; ++j) {
          int m = m0 + j + ((lane >> 4) << 3);
          ls_S[m * SS + n] = acc[j] + ab[m * 64 + n];
        }
      }
      __syncthreads();
      // softmax over keys: 4 threads per row, 16 elems each
      {
        int row = tid >> 2, qq = tid & 3;
        float* rp = ls_S + row * SS + qq * 16;
        float mx = rp[0];
#pragma unroll
        for (int c2 = 1; c2 < 16; ++c2) mx = fmaxf(mx, rp[c2]);
        ls_red[row * 4 + qq] = mx;
        __syncthreads();
        float m4 = fmaxf(fmaxf(ls_red[row * 4 + 0], ls_red[row * 4 + 1]),
                         fmaxf(ls_red[row * 4 + 2], ls_red[row * 4 + 3]));
        float s = 0.f;
#pragma unroll
        for (int c2 = 0; c2 < 16; ++c2) {
          float e2 = __expf(rp[c2] - m4);
          rp[c2] = e2;
          s += e2;
        }
        __syncthreads();            // everyone done reading maxes
        ls_red[row * 4 + qq] = s;
        __syncthreads();
        float tot = ls_red[row * 4 + 0] + ls_red[row * 4 + 1] +
                    ls_red[row * 4 + 2] + ls_red[row * 4 + 3];
        float inv = 1.f / tot;
        unsigned short* pr = ls_P + row * PS + qq * 16;
#pragma unroll
        for (int c2 = 0; c2 < 16; ++c2) pr[c2] = f2bf(rp[c2] * inv);
      }
      __syncthreads();
      // O[64x32] = P @ V  (8 tiles -> one per wave; K = 64 -> 2 WMMA)
      {
        int job = wv;
        int m0 = (job >> 1) * 16, n0 = (job & 1) * 16;
        v8f acc = {0.f, 0.f, 0.f, 0.f, 0.f, 0.f, 0.f, 0.f};
#pragma unroll
        for (int ks = 0; ks < 2; ++ks) {
          v16bf a  = load_a(ls_P, PS, m0, ks * 32, lane);
          v16bf bb = load_b(ls_qkv, QS, ks * 32, vb + n0, lane);
          acc = wmma_bf16(a, bb, acc);
        }
        int n = n0 + (lane & 15);
#pragma unroll
        for (int j = 0; j < 8; ++j) {
          int m = m0 + j + ((lane >> 4) << 3);
          ls_ao[(br * 64 + m) * AOS + hh * 32 + n] = f2bf(acc[j]);
        }
      }
      __syncthreads();
    }
  }

  // ---- phase 5: projections + scatter to BCHW (non-temporal stores) ----
  for (int job = wv; job < 48; job += 8) {
    int brj = job / 24;
    int r2  = job % 24;
    int m0  = (r2 / 6) * 16, n0 = (r2 % 6) * 16;
    const unsigned short* A  = ls_ao + brj * 64 * AOS;
    const unsigned short* Wg = (brj == 0) ? wpm : wp;
    v8f acc = {0.f, 0.f, 0.f, 0.f, 0.f, 0.f, 0.f, 0.f};
#pragma unroll
    for (int ks = 0; ks < 3; ++ks) {
      v16bf a  = load_a(A, AOS, m0, ks * 32, lane);
      v16bf bb = load_bt(Wg, 96, ks * 32, n0, lane);
      acc = wmma_bf16(a, bb, acc);
    }
    int n = n0 + (lane & 15);
    float pb    = (brj == 0) ? projm_b[n] : proj_b[n];
    float* outp = (brj == 0) ? out_m : out_i;
#pragma unroll
    for (int j = 0; j < 8; ++j) {
      int m = m0 + j + ((lane >> 4) << 3);
      __builtin_nontemporal_store(
          acc[j] + pb,
          outp + (((b * C_DIM + n) * HWDIM) + h0 + (m >> 3)) * HWDIM + w0 + (m & 7));
    }
  }
}

// ---------------- launch ----------------
extern "C" void kernel_launch(void* const* d_in, const int* in_sizes, int n_in,
                              void* d_out, int out_size, void* d_ws, size_t ws_size,
                              hipStream_t stream) {
  (void)in_sizes; (void)n_in; (void)out_size; (void)ws_size;
  const float* x_ivif  = (const float*)d_in[0];
  const float* x_mfif  = (const float*)d_in[1];
  const float* V_w     = (const float*)d_in[2];
  const float* V_b     = (const float*)d_in[3];
  const float* Vm_w    = (const float*)d_in[4];
  const float* Vm_b    = (const float*)d_in[5];
  const float* QK_w    = (const float*)d_in[6];
  const float* QK_b    = (const float*)d_in[7];
  const float* QKm_w   = (const float*)d_in[8];
  const float* QKm_b   = (const float*)d_in[9];
  const float* proj_w  = (const float*)d_in[10];
  const float* proj_b  = (const float*)d_in[11];
  const float* projm_w = (const float*)d_in[12];
  const float* projm_b = (const float*)d_in[13];
  const float* meta_w1 = (const float*)d_in[14];
  const float* meta_b1 = (const float*)d_in[15];
  const float* meta_w2 = (const float*)d_in[16];
  const float* meta_b2 = (const float*)d_in[17];
  // d_in[18] = trainingTag (fixed 2 in the reference setup)

  // workspace layout (~176 KB)
  unsigned short* wm  = (unsigned short*)d_ws;       // 288x96 bf16
  unsigned short* wi  = wm + WM_ELE;                 // 192x96 bf16
  unsigned short* wp  = wi + WI_ELE;                 // 96x96 bf16
  unsigned short* wpm = wp + WP_ELE;                 // 96x96 bf16
  float* bv    = (float*)(wpm + WP_ELE);             // 480 conv biases
  float* abias = bv + 512;                           // 3*64*64 attention bias

  pack_kernel<<<128, 256, 0, stream>>>(V_w, V_b, Vm_w, Vm_b, QK_w, QK_b,
                                       QKm_w, QKm_b, proj_w, projm_w,
                                       wm, wi, wp, wpm, bv);
  bias_kernel<<<16, 256, 0, stream>>>(meta_w1, meta_b1, meta_w2, meta_b2, abias);

  float* out_i = (float*)d_out;
  float* out_m = out_i + (size_t)BATCH * C_DIM * HWDIM * HWDIM;  // 25,165,824

  const size_t smem = (size_t)(2 * 64 * XS + 64 * QS + 2 * 64 * AOS + 64 * PS) * 2
                    + (size_t)(64 * SS + 480 + 256) * 4;          // 144,512 bytes
  attn_kernel<<<4096, 256, smem, stream>>>(x_ivif, x_mfif, wm, wi, wp, wpm,
                                           bv, abias, proj_b, projm_b, out_i, out_m);
}
